// GCN_16157666968391
// MI455X (gfx1250) — compile-verified, matrix operand order checked by reading, not run on hardware
//
#include <hip/hip_runtime.h>
#include <hip/hip_bf16.h>

#define NN 50000
#define NE 800000
#define DIM 128
#define NG 256
#define DOUT 10

typedef __attribute__((ext_vector_type(16))) __bf16       v16bf;
typedef __attribute__((ext_vector_type(8)))  float        v8f;
typedef __attribute__((ext_vector_type(4)))  unsigned int u32x4;

union Frag16 { u32x4 q[2]; v16bf v; };

static __device__ __forceinline__ unsigned short f2bf_u(float f) {
  unsigned u = __builtin_bit_cast(unsigned, f);
  u += 0x7FFFu + ((u >> 16) & 1u);            // round-to-nearest-even
  return (unsigned short)(u >> 16);
}
static __device__ __forceinline__ __bf16 f2bf(float f) {
  unsigned short s = f2bf_u(f);
  return __builtin_bit_cast(__bf16, s);
}

__global__ void k_fill(float* p, float v, long long n) {
  long long i = (long long)blockIdx.x * blockDim.x + threadIdx.x;
  if (i < n) p[i] = v;
}

__global__ void k_deg_accum(const long long* __restrict__ dst, float* __restrict__ deg) {
  int e = blockIdx.x * blockDim.x + threadIdx.x;
  if (e < NE)
    __hip_atomic_fetch_add(&deg[dst[e]], 1.0f, __ATOMIC_RELAXED, __HIP_MEMORY_SCOPE_AGENT);
}

__global__ void k_rsqrt(float* d) {
  int i = blockIdx.x * blockDim.x + threadIdx.x;
  if (i < NN) d[i] = rsqrtf(d[i]);   // deg >= 1 always (self-loops)
}

__global__ void k_bias_init(float* __restrict__ h, const float* __restrict__ b, long long n) {
  long long i = (long long)blockIdx.x * blockDim.x + threadIdx.x;
  if (i < n) h[i] = b[i & (DIM - 1)];
}

// f32 -> bf16 row-major, two elements per thread (one packed dword store).
__global__ void k_cvt_bf16(const float* __restrict__ src, unsigned* __restrict__ dst,
                           long long npairs) {
  long long i = (long long)blockIdx.x * blockDim.x + threadIdx.x;
  if (i >= npairs) return;
  const float2 f = ((const float2*)src)[i];
  dst[i] = ((unsigned)f2bf_u(f.y) << 16) | (unsigned)f2bf_u(f.x);
}

// Pack f32 weight B[K][N] into per-lane WMMA B-fragment layout:
// flat index t = (ntile*(K/32) + kt)*32 + lane ; Bp[t*16 .. t*16+15] = lane's 16 bf16.
__global__ void k_pack_B(const float* __restrict__ B, __bf16* __restrict__ Bp,
                         int K, int N) {
  const int t = blockIdx.x * blockDim.x + threadIdx.x;
  const int total = (K / 32) * (N / 16) * 32;
  if (t >= total) return;
  const int lane  = t & 31;
  const int rest  = t >> 5;
  const int kt    = rest % (K / 32);
  const int ntile = rest / (K / 32);
  const int hh  = lane >> 4;
  const int n   = ntile * 16 + (lane & 15);
  __bf16* out = Bp + (size_t)t * 16;
#pragma unroll
  for (int j = 0; j < 8; ++j) {
    const int k = kt * 32 + ((j < 4) ? (8 * hh + 2 * j) : (16 + 8 * hh + 2 * (j - 4)));
    out[2 * j]     = f2bf(B[(size_t)k * N + n]);
    out[2 * j + 1] = f2bf(B[(size_t)(k + 1) * N + n]);
  }
}

// C[M,N] = A[M,KK](bf16,row-major) @ Bp(packed bf16) [+bias, relu].
// grid = (M/16, N/128), block = 256 (8 waves, one 16x16 C tile each).
template <int KK, bool BR>
__global__ __launch_bounds__(256) void k_gemm(
    const __bf16* __restrict__ A, const __bf16* __restrict__ Bp,
    float* __restrict__ C, int N, const float* __restrict__ bias) {
  const int wave  = threadIdx.x >> 5;
  const int lane  = threadIdx.x & 31;
  const int l15   = lane & 15;
  const int hh    = lane >> 4;
  const int mbase = blockIdx.x * 16;
  const int ntile = blockIdx.y * 8 + wave;
  const int nbase = ntile * 16;
  const __bf16* arow = A + (size_t)(mbase + l15) * KK;

  v8f c = {};
#pragma unroll
  for (int kt = 0; kt < KK / 32; ++kt) {
    Frag16 a, b;
    const int kb = kt * 32;
    // A 16-bit fragment: two contiguous 8-element (16B) runs per lane
    a.q[0] = *(const u32x4*)(arow + kb + 8 * hh);
    a.q[1] = *(const u32x4*)(arow + kb + 16 + 8 * hh);
    const __bf16* bp = Bp + ((size_t)(ntile * (KK / 32) + kt) * 32 + lane) * 16;
    b.q[0] = *(const u32x4*)(bp);
    b.q[1] = *(const u32x4*)(bp + 8);
    c = __builtin_amdgcn_wmma_f32_16x16x32_bf16(false, a.v, false, b.v, (short)0, c,
                                                false, false);
  }

  float bv = 0.0f;
  if (BR) bv = bias[nbase + l15];
#pragma unroll
  for (int r = 0; r < 8; ++r) {
    const int mr = mbase + r + 8 * hh;   // C layout: VGPR r, half-wave selects M+8
    float v = c[r];
    if (BR) { v += bv; v = v > 0.0f ? v : 0.0f; }
    C[(size_t)mr * N + nbase + l15] = v;
  }
}

// One wave per (edge or self-loop); lane covers 4 consecutive features.
__global__ __launch_bounds__(256) void k_scatter(
    const float* __restrict__ hw, const long long* __restrict__ src,
    const long long* __restrict__ dst, const float* __restrict__ dinv,
    float* __restrict__ hout) {
  const long long wid = ((long long)blockIdx.x * blockDim.x + threadIdx.x) >> 5;
  const int lane = threadIdx.x & 31;
  if (wid >= (long long)NE + NN) return;
  long long s, d;
  if (wid < NE) { s = src[wid]; d = dst[wid]; }
  else          { s = wid - NE; d = s; }           // self-loop
  const float nrm = dinv[s] * dinv[d];
  const float4 v = ((const float4*)(hw + s * DIM))[lane];
  float* out = hout + d * DIM + lane * 4;
  __hip_atomic_fetch_add(out + 0, v.x * nrm, __ATOMIC_RELAXED, __HIP_MEMORY_SCOPE_AGENT);
  __hip_atomic_fetch_add(out + 1, v.y * nrm, __ATOMIC_RELAXED, __HIP_MEMORY_SCOPE_AGENT);
  __hip_atomic_fetch_add(out + 2, v.z * nrm, __ATOMIC_RELAXED, __HIP_MEMORY_SCOPE_AGENT);
  __hip_atomic_fetch_add(out + 3, v.w * nrm, __ATOMIC_RELAXED, __HIP_MEMORY_SCOPE_AGENT);
}

// global_add_pool into concatenated g[NG][2*DIM] at column offset.
__global__ void k_pool(const float* __restrict__ h, const long long* __restrict__ batch,
                       float* __restrict__ g, int offset) {
  const long long i = (long long)blockIdx.x * blockDim.x + threadIdx.x;
  if (i >= (long long)NN * DIM) return;
  const int nd = (int)(i >> 7);
  const int dd = (int)(i & (DIM - 1));
  const long long gb = batch[nd];
  __hip_atomic_fetch_add(&g[gb * (2 * DIM) + offset + dd], h[i],
                         __ATOMIC_RELAXED, __HIP_MEMORY_SCOPE_AGENT);
}

// Final [256,256]@[256,10]+bl2, then h and log_softmax(h) -> d_out.
__global__ void k_head(const float* __restrict__ gh, const float* __restrict__ Wl2,
                       const float* __restrict__ bl2, float* __restrict__ out) {
  const int r = blockIdx.x * blockDim.x + threadIdx.x;
  if (r >= NG) return;
  float acc[DOUT];
#pragma unroll
  for (int c2 = 0; c2 < DOUT; ++c2) acc[c2] = bl2[c2];
  for (int k = 0; k < 2 * DIM; ++k) {
    const float gv = gh[r * (2 * DIM) + k];
#pragma unroll
    for (int c2 = 0; c2 < DOUT; ++c2) acc[c2] += gv * Wl2[k * DOUT + c2];
  }
  float mx = acc[0];
#pragma unroll
  for (int c2 = 1; c2 < DOUT; ++c2) mx = fmaxf(mx, acc[c2]);
  float se = 0.0f;
#pragma unroll
  for (int c2 = 0; c2 < DOUT; ++c2) se += __expf(acc[c2] - mx);
  const float lse = mx + __logf(se);
#pragma unroll
  for (int c2 = 0; c2 < DOUT; ++c2) {
    out[r * DOUT + c2] = acc[c2];                    // tuple output 0: h
    out[NG * DOUT + r * DOUT + c2] = acc[c2] - lse;  // tuple output 1: log_softmax
  }
}

extern "C" void kernel_launch(void* const* d_in, const int* in_sizes, int n_in,
                              void* d_out, int out_size, void* d_ws, size_t ws_size,
                              hipStream_t stream) {
  const float*     x    = (const float*)d_in[0];
  const long long* eidx = (const long long*)d_in[1];   // [2][NE] int64
  const long long* bat  = (const long long*)d_in[2];   // [NN]    int64
  const float* W1  = (const float*)d_in[3];
  const float* b1  = (const float*)d_in[4];
  const float* W2  = (const float*)d_in[5];
  const float* b2  = (const float*)d_in[6];
  const float* Wl1 = (const float*)d_in[7];
  const float* bl1 = (const float*)d_in[8];
  const float* Wl2 = (const float*)d_in[9];
  const float* bl2 = (const float*)d_in[10];
  const long long* esrc = eidx;
  const long long* edst = eidx + NE;

  char* ws = (char*)d_ws;
  float*  bufA = (float*)ws;  ws += (size_t)NN * DIM * 4;          // x@W1 / h1@W2
  float*  bufB = (float*)ws;  ws += (size_t)NN * DIM * 4;          // h1 / h2
  __bf16* hb   = (__bf16*)ws; ws += (size_t)NN * DIM * 2;          // bf16 activations
  float*  dinv = (float*)ws;  ws += ((size_t)NN * 4 + 255) & ~(size_t)255;
  float*  g    = (float*)ws;  ws += (size_t)NG * 2 * DIM * 4;
  float*  gh   = (float*)ws;  ws += (size_t)NG * 2 * DIM * 4;
  __bf16* gb   = (__bf16*)ws; ws += (size_t)NG * 2 * DIM * 2;
  __bf16* W1p  = (__bf16*)ws; ws += (size_t)DIM * DIM * 2;
  __bf16* W2p  = (__bf16*)ws; ws += (size_t)DIM * DIM * 2;
  __bf16* Wl1p = (__bf16*)ws; ws += (size_t)(2 * DIM) * (2 * DIM) * 2;

  const long long ND = (long long)NN * DIM;
  const int B256 = 256;
  const int scat_blocks = (int)((((long long)NE + NN) * 32 + B256 - 1) / B256);
  const int cvt_blocks  = (int)((ND / 2 + B256 - 1) / B256);

  // degrees -> dinv (deg starts at 1.0 for the self-loop)
  k_fill<<<(NN + B256 - 1) / B256, B256, 0, stream>>>(dinv, 1.0f, NN);
  k_deg_accum<<<(NE + B256 - 1) / B256, B256, 0, stream>>>(edst, dinv);
  k_rsqrt<<<(NN + B256 - 1) / B256, B256, 0, stream>>>(dinv);

  // pack weights into B-fragment layout (tiny)
  k_pack_B<<<4,  B256, 0, stream>>>(W1,  W1p,  DIM,     DIM);
  k_pack_B<<<4,  B256, 0, stream>>>(W2,  W2p,  DIM,     DIM);
  k_pack_B<<<16, B256, 0, stream>>>(Wl1, Wl1p, 2 * DIM, 2 * DIM);

  // layer 1: hb = bf16(x); bufA = hb@W1 ; bufB = b1 + scatter(bufA)
  k_cvt_bf16<<<cvt_blocks, B256, 0, stream>>>(x, (unsigned*)hb, ND / 2);
  k_gemm<DIM, false><<<dim3(NN / 16, 1), B256, 0, stream>>>(hb, W1p, bufA, DIM, nullptr);
  k_bias_init<<<(int)((ND + B256 - 1) / B256), B256, 0, stream>>>(bufB, b1, ND);
  k_scatter<<<scat_blocks, B256, 0, stream>>>(bufA, esrc, edst, dinv, bufB);

  // pool h1 -> g[:, 0:128]
  k_fill<<<(NG * 2 * DIM + B256 - 1) / B256, B256, 0, stream>>>(g, 0.0f, NG * 2 * DIM);
  k_pool<<<(int)((ND + B256 - 1) / B256), B256, 0, stream>>>(bufB, bat, g, 0);

  // layer 2: hb = bf16(h1); bufA = hb@W2 ; bufB = b2 + scatter(bufA)
  k_cvt_bf16<<<cvt_blocks, B256, 0, stream>>>(bufB, (unsigned*)hb, ND / 2);
  k_gemm<DIM, false><<<dim3(NN / 16, 1), B256, 0, stream>>>(hb, W2p, bufA, DIM, nullptr);
  k_bias_init<<<(int)((ND + B256 - 1) / B256), B256, 0, stream>>>(bufB, b2, ND);
  k_scatter<<<scat_blocks, B256, 0, stream>>>(bufA, esrc, edst, dinv, bufB);

  // pool h2 -> g[:, 128:256]
  k_pool<<<(int)((ND + B256 - 1) / B256), B256, 0, stream>>>(bufB, bat, g, DIM);

  // head: gh = relu(gb@Wl1 + bl1) ; logits + log_softmax -> d_out
  k_cvt_bf16<<<(NG * 2 * DIM / 2 + B256 - 1) / B256, B256, 0, stream>>>(g, (unsigned*)gb,
                                                                        NG * 2 * DIM / 2);
  k_gemm<2 * DIM, true><<<dim3(NG / 16, 2), B256, 0, stream>>>(gb, Wl1p, gh, 2 * DIM, bl1);
  k_head<<<1, NG, 0, stream>>>(gh, Wl2, bl2, (float*)d_out);
}
